// PairPotentials_58076547777143
// MI455X (gfx1250) — matrix operand order repaired and checked by reading, not compiled, not run on hardware
//
#include <hip/hip_runtime.h>

typedef __attribute__((ext_vector_type(16))) _Float16 v16h;
typedef __attribute__((ext_vector_type(8)))  float    v8f;

#define WMMA_F16(a, b, c) \
  __builtin_amdgcn_wmma_f32_16x16x32_f16(false, (a), false, (b), (short)0, (c), false, false)

// One wave (32 lanes) processes 32 pairs per iteration:
//  - lane L computes d for pair base+L (coalesced index loads, L2-resident xyz gathers)
//  - two 16-pair M-blocks; per block, lanes L and L+16 build disjoint K-halves of the
//    16x32 f16 A-tile of tanh activations (ISA 7.12.2 layout), two K-chunks cover HIDDEN=64
//  - B = W2 chunk broadcast across all 16 columns -> D[m][*] = per-pair energy (f32 accum)
__global__ __launch_bounds__(256)
void PairPotentials_pair_energy(const float* __restrict__ xyz,
                                const float* __restrict__ cell,
                                const int*   __restrict__ nbr,
                                const int*   __restrict__ off,
                                const float* __restrict__ W1,
                                const float* __restrict__ b1v_g,
                                const float* __restrict__ W2,
                                float*       __restrict__ partials,
                                int nPairs)
{
    const int lane        = threadIdx.x & 31;
    const int wid         = threadIdx.x >> 5;
    const int half        = lane >> 4;                 // which lane-half (K layout)
    const int wavesPerBlk = blockDim.x >> 5;
    const int waveId      = blockIdx.x * wavesPerBlk + wid;
    const int stride      = (gridDim.x * wavesPerBlk) * 32;

    // loop-invariant 3x3 cell (uniform -> scalar loads)
    const float c00 = cell[0], c01 = cell[1], c02 = cell[2];
    const float c10 = cell[3], c11 = cell[4], c12 = cell[5];
    const float c20 = cell[6], c21 = cell[7], c22 = cell[8];

    // Per-lane W1/b1 slices in the 16-bit A VGPR layout, and B fragments (W2
    // replicated on all 16 columns) in the 32x16 16-bit B layout.
    float w1v[32], b1v[32];
    v16h  bf0, bf1;
#pragma unroll
    for (int e = 0; e < 16; ++e) {
        const int k = 16 * (e >> 3) + 8 * half + 2 * ((e >> 1) & 3) + (e & 1);
        w1v[e]      = W1[k];        b1v[e]      = b1v_g[k];
        w1v[16 + e] = W1[32 + k];   b1v[16 + e] = b1v_g[32 + k];
        const int kb = 16 * (e >> 3) + 8 * half + 2 * ((e >> 1) & 3) + (e & 1);
        bf0[e] = (_Float16)W2[kb];
        bf1[e] = (_Float16)W2[32 + kb];
    }

    v8f acc = {};  // f32 accumulator, chained across the whole loop

    for (int base = waveId * 32; base < nPairs; base += stride) {
        const int p  = base + lane;
        const int pc = (p < nPairs) ? p : (nPairs - 1);   // clamp: keeps loads in-bounds

        // prefetch next iteration's index streams (global_prefetch_b8, speculative)
        {
            int pn = p + stride; pn = (pn < nPairs) ? pn : (nPairs - 1);
            __builtin_prefetch(&nbr[2 * pn], 0, 1);
            __builtin_prefetch(&off[3 * pn], 0, 1);
        }

        const int i0 = nbr[2 * pc + 0];
        const int j0 = nbr[2 * pc + 1];
        const float o0 = (float)off[3 * pc + 0];
        const float o1 = (float)off[3 * pc + 1];
        const float o2 = (float)off[3 * pc + 2];
        const float* ri = xyz + 3 * i0;
        const float* rj = xyz + 3 * j0;
        const float dx = ri[0] - rj[0] + o0 * c00 + o1 * c10 + o2 * c20;
        const float dy = ri[1] - rj[1] + o0 * c01 + o1 * c11 + o2 * c21;
        const float dz = ri[2] - rj[2] + o0 * c02 + o1 * c12 + o2 * c22;
        const float dl = sqrtf(dx * dx + dy * dy + dz * dz);

#pragma unroll
        for (int mblk = 0; mblk < 2; ++mblk) {
            const int   src = mblk * 16 + (lane & 15);   // A row m = lane&15
            const float d   = __shfl(dl, src, 32);
            const float msk = ((base + src) < nPairs) ? 1.0f : 0.0f;  // zero tail rows
            v16h a0, a1;
#pragma unroll
            for (int e = 0; e < 16; ++e) {
                a0[e] = (_Float16)(msk * tanhf(fmaf(w1v[e],      d, b1v[e])));
                a1[e] = (_Float16)(msk * tanhf(fmaf(w1v[16 + e], d, b1v[16 + e])));
            }
            acc = WMMA_F16(a0, bf0, acc);   // K = 0..31
            acc = WMMA_F16(a1, bf1, acc);   // K = 32..63
        }
    }

    // D[m][n] is independent of n; lanes 0-15 hold rows M=0..7, lanes 16-31 rows M=8..15.
    const float s = acc[0] + acc[1] + acc[2] + acc[3] + acc[4] + acc[5] + acc[6] + acc[7];
    const float wave_sum = __shfl(s, 0, 32) + __shfl(s, 16, 32);

    __shared__ float lds[32];
    if (lane == 0) lds[wid] = wave_sum;
    __syncthreads();
    if (threadIdx.x == 0) {
        float t = 0.0f;
        for (int w = 0; w < wavesPerBlk; ++w) t += lds[w];
        partials[blockIdx.x] = t;
    }
}

// Deterministic fixed-order reduction of block partials + the N_PAIRS * b2 bias term.
__global__ __launch_bounds__(256)
void PairPotentials_reduce(const float* __restrict__ partials, int n,
                           const float* __restrict__ b2, int nPairs,
                           float* __restrict__ out)
{
    __shared__ float lds[256];
    float t = 0.0f;
    for (int i = threadIdx.x; i < n; i += 256) t += partials[i];
    lds[threadIdx.x] = t;
    __syncthreads();
    for (int s = 128; s > 0; s >>= 1) {
        if (threadIdx.x < s) lds[threadIdx.x] += lds[threadIdx.x + s];
        __syncthreads();
    }
    if (threadIdx.x == 0) out[0] = lds[0] + b2[0] * (float)nPairs;
}

extern "C" void kernel_launch(void* const* d_in, const int* in_sizes, int n_in,
                              void* d_out, int out_size, void* d_ws, size_t ws_size,
                              hipStream_t stream)
{
    const float* xyz  = (const float*)d_in[0];
    const float* cell = (const float*)d_in[1];
    const int*   nbr  = (const int*)  d_in[2];
    const int*   off  = (const int*)  d_in[3];
    const float* W1   = (const float*)d_in[4];
    const float* b1   = (const float*)d_in[5];
    const float* W2   = (const float*)d_in[6];
    const float* b2   = (const float*)d_in[7];
    float* out = (float*)d_out;

    const int nPairs = in_sizes[2] / 2;   // nbr_list is (N_PAIRS, 2)

    int nblocks = 1024;                   // 8 waves/block * 1024 blocks: plenty of latency hiding
    if (ws_size < (size_t)nblocks * sizeof(float)) {
        nblocks = (int)(ws_size / sizeof(float));
        if (nblocks < 1) nblocks = 1;
    }
    float* partials = (float*)d_ws;

    PairPotentials_pair_energy<<<nblocks, 256, 0, stream>>>(
        xyz, cell, nbr, off, W1, b1, W2, partials, nPairs);
    PairPotentials_reduce<<<1, 256, 0, stream>>>(partials, nblocks, b2, nPairs, out);
}
